// CTCPredNetV1_15221364097065
// MI455X (gfx1250) — matrix operand order repaired and checked by reading, not compiled, not run on hardware
//
#include <hip/hip_runtime.h>
#include <hip/hip_bf16.h>
#include <math.h>

typedef __attribute__((ext_vector_type(16))) __bf16 bf16x16;
typedef __attribute__((ext_vector_type(8)))  float  f32x8;
typedef __attribute__((ext_vector_type(4)))  unsigned u32x4;
typedef int i32x4v __attribute__((vector_size(16)));   // matches builtin pointee type
typedef unsigned short u16;

#define B_    32
#define L_    1000
#define IN_   80
#define INP_  96      // IN padded to mult of 32
#define HID_  256
#define KVQ_  512
#define NCLS_ 72
#define GATE_ 1024    // 4*HID

// CDNA5 async global->LDS copy path (ASYNCcnt) with portable fallback.
#if defined(__HIP_DEVICE_COMPILE__) && __has_builtin(__builtin_amdgcn_global_load_async_to_lds_b128)
#define HAVE_ASYNC_LDS 1
#else
#define HAVE_ASYNC_LDS 0
#endif

// ---------------- helpers ----------------

__device__ __forceinline__ u16 f2bf(float f) {
  unsigned u = __builtin_bit_cast(unsigned, f);
  u += 0x7fffu + ((u >> 16) & 1u);              // round to nearest even
  return (u16)(u >> 16);
}

union FragAB { unsigned u[8]; bf16x16 v; };
union FragC  { float f[8];    f32x8   v; };

__device__ __forceinline__ f32x8 wmma_bf16(bf16x16 a, bf16x16 b, f32x8 c) {
  return __builtin_amdgcn_wmma_f32_16x16x32_bf16(false, a, false, b, (short)0, c,
                                                 false, false);
}

// 16-byte global -> LDS copy. Async (ASYNCcnt) on CDNA5, sync fallback otherwise.
__device__ __forceinline__ void cp_g2l_16(const u16* gsrc, u16* ldst) {
#if HAVE_ASYNC_LDS
  __builtin_amdgcn_global_load_async_to_lds_b128(
      (__attribute__((address_space(1))) i32x4v*)gsrc,
      (__attribute__((address_space(3))) i32x4v*)ldst, 0, 0);
#else
  *(u32x4*)ldst = *(const u32x4*)gsrc;
#endif
}

__device__ __forceinline__ void cp_g2l_wait() {
#if HAVE_ASYNC_LDS
#if __has_builtin(__builtin_amdgcn_s_wait_asynccnt)
  __builtin_amdgcn_s_wait_asynccnt(0);
#else
  asm volatile("s_wait_asynccnt 0x0" ::: "memory");
#endif
#endif
}

// A fragment: 16x32 bf16 tile at (m0,k0) from row-major [*, lda]; clamp m<M.
// Lane l: m = m0+(l&15); VGPR v covers k = k0 + (l>>4)*16 + v*2.
__device__ __forceinline__ bf16x16 ldfrag_A(const u16* A, long lda, int m0, int k0, int M) {
  int l = threadIdx.x & 31;
  int m = m0 + (l & 15);
  if (m >= M) m = M - 1;
  const unsigned* row = (const unsigned*)(A + (long)m * lda);
  int kh = (l >> 4) * 16;
  FragAB f;
#pragma unroll
  for (int v = 0; v < 8; ++v) {
    int k = k0 + kh + v * 2;
    f.u[v] = row[k >> 1];
  }
  return f.v;
}

// B fragment (KxN) for NT: B stored [N,K] row-major; clamp n<N.
// Lane l: n = n0+(l&15); VGPR v covers k = k0 + (l>>4)*16 + v*2.
__device__ __forceinline__ bf16x16 ldfrag_B_nt(const u16* Bm, long ldb, int n0, int k0, int N) {
  int l = threadIdx.x & 31;
  int n = n0 + (l & 15);
  if (n >= N) n = N - 1;
  const unsigned* row = (const unsigned*)(Bm + (long)n * ldb);
  int kb = (k0 + (l >> 4) * 16) >> 1;
  FragAB f;
#pragma unroll
  for (int v = 0; v < 8; ++v) f.u[v] = row[kb + v];
  return f.v;
}

// B fragment for NN: B stored [K,N] row-major; clamp k<Kreal.
__device__ __forceinline__ bf16x16 ldfrag_B_nn(const u16* Bm, long ldb, int n0, int k0,
                                               int Kreal) {
  int l = threadIdx.x & 31;
  int n = n0 + (l & 15);
  int kb = k0 + (l >> 4) * 16;
  FragAB f;
#pragma unroll
  for (int v = 0; v < 8; ++v) {
    int ke0 = kb + v * 2, ke1 = ke0 + 1;
    if (ke0 >= Kreal) ke0 = Kreal - 1;
    if (ke1 >= Kreal) ke1 = Kreal - 1;
    unsigned lo = Bm[(long)ke0 * ldb + n];
    unsigned hi = Bm[(long)ke1 * ldb + n];
    f.u[v] = lo | (hi << 16);
  }
  return f.v;
}

// ---------------- LDS-staged, double-buffered WMMA GEMMs ----------------
// C[M,N] = scale*(A[M,Kp] @ B^T) + bias ; optional relu; f32 and/or bf16 out.
// block = 256 thr (8 waves). block tile 128(M) x 64(N), K-step 32, double-buffered
// LDS (async global->LDS fills). grid (ceil(N/64), ceil(M/128), batch).

#define AST 40   // padded LDS row stride (bf16): 16 rows hit 16 distinct banks

__global__ void __launch_bounds__(256)
gemm_nt(const u16* __restrict__ A, const u16* __restrict__ Bm,
        float* Cf, u16* Cb, const float* bias,
        int M, int N, int Kp, long lda, long ldb, long ldc,
        long strA, long strB, long strC, float scale, int relu) {
  long bz = blockIdx.z;
  A += bz * strA; Bm += bz * strB;
  int wid = threadIdx.x >> 5;
  int m0b = blockIdx.y * 128;
  int n0b = blockIdx.x * 64;

  __shared__ u16 As[2][128 * AST];   // 20.0 KB
  __shared__ u16 Bs[2][64 * AST];    // 10.0 KB

  int t = threadIdx.x;
  int rA = t >> 1, sA = (t & 1) * 2;       // A: 128 rows x 2 thr, 2x16B each
  int gmA = m0b + rA; if (gmA >= M) gmA = M - 1;
  const u16* gA = A + (long)gmA * lda;
  int rB = t >> 2, sB = t & 3;             // B: 64 rows x 4 thr, 1x16B each
  int gnB = n0b + rB; if (gnB >= N) gnB = N - 1;
  const u16* gB = Bm + (long)gnB * ldb;

  auto stage = [&](int buf, int k0) {
    cp_g2l_16(gA + k0 + sA * 8,       &As[buf][rA * AST + sA * 8]);
    cp_g2l_16(gA + k0 + (sA + 1) * 8, &As[buf][rA * AST + (sA + 1) * 8]);
    cp_g2l_16(gB + k0 + sB * 8,       &Bs[buf][rB * AST + sB * 8]);
  };

  int ntile = Kp >> 5;
  stage(0, 0);
  cp_g2l_wait();
  __syncthreads();

  f32x8 acc[4] = {};
  int buf = 0;
  for (int it = 0; it < ntile; ++it) {
    if (it + 1 < ntile) stage(buf ^ 1, (it + 1) * 32);
    bf16x16 a = ldfrag_A(As[buf], AST, wid * 16, 0, 1 << 30);
#pragma unroll
    for (int i = 0; i < 4; ++i) {
      bf16x16 b = ldfrag_B_nt(Bs[buf], AST, i * 16, 0, 1 << 30);
      acc[i] = wmma_bf16(a, b, acc[i]);
    }
    cp_g2l_wait();
    __syncthreads();
    buf ^= 1;
  }

  int l = threadIdx.x & 31, nl = l & 15, mh = (l >> 4) * 8;
  int m0 = m0b + wid * 16;
#pragma unroll
  for (int i = 0; i < 4; ++i) {
    int n = n0b + i * 16 + nl;
    if (n >= N) continue;
    float bv = bias ? bias[n] : 0.f;
    FragC c; c.v = acc[i];
#pragma unroll
    for (int v = 0; v < 8; ++v) {
      int m = m0 + mh + v;
      if (m < M) {
        float r = c.f[v] * scale + bv;
        if (relu) r = fmaxf(r, 0.f);
        long idx = (long)m * ldc + n + bz * strC;
        if (Cf) Cf[idx] = r;
        if (Cb) Cb[idx] = f2bf(r);
      }
    }
  }
}

#define BSTN 72  // padded LDS row stride for NN B tile (32 K-rows x 64 N-cols)

__global__ void __launch_bounds__(256)
gemm_nn(const u16* __restrict__ A, const u16* __restrict__ Bm,
        float* Cf, u16* Cb, const float* bias,
        int M, int N, int Kp, int Kreal, long lda, long ldb, long ldc,
        long strA, long strB, long strC, float scale, int relu) {
  // NOTE: requires N multiple of 64 (holds for all call sites).
  long bz = blockIdx.z;
  A += bz * strA; Bm += bz * strB;
  int wid = threadIdx.x >> 5;
  int m0b = blockIdx.y * 128;
  int n0b = blockIdx.x * 64;

  __shared__ u16 As[2][128 * AST];   // 20.0 KB
  __shared__ u16 Bs[2][32 * BSTN];   // 9.0 KB

  int t = threadIdx.x;
  int rA = t >> 1, sA = (t & 1) * 2;
  int gmA = m0b + rA; if (gmA >= M) gmA = M - 1;
  const u16* gA = A + (long)gmA * lda;
  int rB = t >> 3, sB = t & 7;             // B: 32 K-rows x 8 thr, 1x16B each

  auto stage = [&](int buf, int k0) {
    cp_g2l_16(gA + k0 + sA * 8,       &As[buf][rA * AST + sA * 8]);
    cp_g2l_16(gA + k0 + (sA + 1) * 8, &As[buf][rA * AST + (sA + 1) * 8]);
    int gk = k0 + rB; if (gk >= Kreal) gk = Kreal - 1;
    cp_g2l_16(Bm + (long)gk * ldb + n0b + sB * 8, &Bs[buf][rB * BSTN + sB * 8]);
  };

  int ntile = Kp >> 5;
  stage(0, 0);
  cp_g2l_wait();
  __syncthreads();

  f32x8 acc[4] = {};
  int buf = 0;
  for (int it = 0; it < ntile; ++it) {
    if (it + 1 < ntile) stage(buf ^ 1, (it + 1) * 32);
    bf16x16 a = ldfrag_A(As[buf], AST, wid * 16, 0, 1 << 30);
#pragma unroll
    for (int i = 0; i < 4; ++i) {
      bf16x16 b = ldfrag_B_nn(Bs[buf], BSTN, i * 16, 0, 32);
      acc[i] = wmma_bf16(a, b, acc[i]);
    }
    cp_g2l_wait();
    __syncthreads();
    buf ^= 1;
  }

  int l = threadIdx.x & 31, nl = l & 15, mh = (l >> 4) * 8;
  int m0 = m0b + wid * 16;
#pragma unroll
  for (int i = 0; i < 4; ++i) {
    int n = n0b + i * 16 + nl;
    if (n >= N) continue;
    float bv = bias ? bias[n] : 0.f;
    FragC c; c.v = acc[i];
#pragma unroll
    for (int v = 0; v < 8; ++v) {
      int m = m0 + mh + v;
      if (m < M) {
        float r = c.f[v] * scale + bv;
        if (relu) r = fmaxf(r, 0.f);
        long idx = (long)m * ldc + n + bz * strC;
        if (Cf) Cf[idx] = r;
        if (Cb) Cb[idx] = f2bf(r);
      }
    }
  }
}

// ---------------- LSTM (sequential scan, one WG per direction) ----------------
// 1024 threads = 32 waves. Wave (mt,jt) owns gate tiles for batch rows mt*16..+15
// and hidden cols jt*16..+15 across ALL four gates -> cell state c stays in VGPRs.
// g = [h ; x_t] @ [Whh ; Wih]^T + b, K=512 per tile, bf16 WMMA; weights stream
// from L2 (1MB/step/dir working set cannot fit 320KB LDS).

__global__ void __launch_bounds__(1024)
lstm_scan(const u16* __restrict__ x_bf,                       // [B*L, 256]
          const u16* WihF, const u16* WhhF, const float* bF,  // fwd
          const u16* WihB, const u16* WhhB, const float* bB,  // bwd
          u16* __restrict__ hcat_bf) {                        // [B*L, 512]
  int dir = blockIdx.x;
  const u16* Wih = dir ? WihB : WihF;
  const u16* Whh = dir ? WhhB : WhhF;
  const float* bb = dir ? bB : bF;

  __shared__ u16 h_lds[B_ * HID_];   // 32x256 bf16 = 16KB

  int tid = threadIdx.x;
  for (int i = tid; i < B_ * HID_; i += blockDim.x) h_lds[i] = 0;
  __syncthreads();

  int wid = tid >> 5, l = tid & 31;
  int mt = wid >> 4, jt = wid & 15;
  int m0 = mt * 16, j0 = jt * 16;
  int nl = l & 15, mh = (l >> 4) * 8;

  float bi[4];
#pragma unroll
  for (int g = 0; g < 4; ++g) bi[g] = bb[g * HID_ + j0 + nl];

  float creg[8] = {0.f};

  const unsigned* xrow_base =
      (const unsigned*)(x_bf + ((long)(m0 + nl) * L_) * HID_);
  int kh = (l >> 4) * 16;

  for (int t = 0; t < L_; ++t) {
    int te = dir ? (L_ - 1 - t) : t;
    f32x8 acc[4] = {};

    // recurrent part: h (LDS) @ Whh^T
#pragma unroll
    for (int ks = 0; ks < 8; ++ks) {
      int k0 = ks * 32;
      bf16x16 a = ldfrag_A(h_lds, HID_, m0, k0, B_);
#pragma unroll
      for (int g = 0; g < 4; ++g) {
        bf16x16 b = ldfrag_B_nt(Whh, HID_, g * HID_ + j0, k0, GATE_);
        acc[g] = wmma_bf16(a, b, acc[g]);
      }
    }
    // input part: x_t (global) @ Wih^T ; row for batch m is (m*L + te)
    const unsigned* xrow = xrow_base + ((long)te * HID_ >> 1);
#pragma unroll
    for (int ks = 0; ks < 8; ++ks) {
      int k0 = ks * 32;
      FragAB fa;
#pragma unroll
      for (int v = 0; v < 8; ++v) {
        int k = k0 + kh + v * 2;
        fa.u[v] = xrow[k >> 1];
      }
#pragma unroll
      for (int g = 0; g < 4; ++g) {
        bf16x16 b = ldfrag_B_nt(Wih, HID_, g * HID_ + j0, k0, GATE_);
        acc[g] = wmma_bf16(fa.v, b, acc[g]);
      }
    }

    FragC gi, gf, gc, go;
    gi.v = acc[0]; gf.v = acc[1]; gc.v = acc[2]; go.v = acc[3];
    u16 hout[8];
#pragma unroll
    for (int v = 0; v < 8; ++v) {
      float iv = 1.f / (1.f + __expf(-(gi.f[v] + bi[0])));
      float fv = 1.f / (1.f + __expf(-(gf.f[v] + bi[1])));
      float gv = tanhf(gc.f[v] + bi[2]);
      float ov = 1.f / (1.f + __expf(-(go.f[v] + bi[3])));
      float c = fv * creg[v] + iv * gv;
      creg[v] = c;
      hout[v] = f2bf(ov * tanhf(c));
    }

    __syncthreads();   // all LDS reads of old h complete
#pragma unroll
    for (int v = 0; v < 8; ++v) {
      int m = m0 + mh + v;
      int j = j0 + nl;
      h_lds[m * HID_ + j] = hout[v];
      hcat_bf[((long)m * L_ + te) * (2 * HID_) + dir * HID_ + j] = hout[v];
    }
    __syncthreads();   // new h visible before next step
  }
}

// ---------------- small elementwise / reduction kernels ----------------

__global__ void cvt_pad(const float* __restrict__ src, u16* __restrict__ dst,
                        int rows, int cols, int colsp) {
  long i = (long)blockIdx.x * blockDim.x + threadIdx.x;
  long total = (long)rows * colsp;
  if (i >= total) return;
  int c = (int)(i % colsp);
  long r = i / colsp;
  float v = (c < cols) ? src[r * (long)cols + c] : 0.f;
  dst[i] = f2bf(v);
}

__global__ void vq_norm_k(const float* __restrict__ emb, float* __restrict__ nrm) {
  int k = blockIdx.x * blockDim.x + threadIdx.x;
  if (k >= KVQ_) return;
  float s = 0.f;
  for (int c = 0; c < HID_; ++c) { float e = emb[(long)k * HID_ + c]; s += e * e; }
  nrm[k] = s;
}

// argmin_k (||e_k||^2 - 2*score) ; gather zq = emb[best]. One wave per row.
__global__ void vq_select(const float* __restrict__ score, const float* __restrict__ nrm,
                          const float* __restrict__ emb, float* __restrict__ zq) {
  int row = blockIdx.x * 8 + (threadIdx.x >> 5);
  int l = threadIdx.x & 31;
  if (row >= B_ * L_) return;
  const float* s = score + (long)row * KVQ_;
  float best = 3.0e38f; int bidx = 0;
  for (int k = l; k < KVQ_; k += 32) {
    float d = nrm[k] - 2.f * s[k];
    if (d < best) { best = d; bidx = k; }
  }
#pragma unroll
  for (int off = 16; off; off >>= 1) {
    float ob = __shfl_xor(best, off, 32);
    int   oi = __shfl_xor(bidx, off, 32);
    if (ob < best || (ob == best && oi < bidx)) { best = ob; bidx = oi; }
  }
  const float* e = emb + (long)bidx * HID_;
  float* o = zq + (long)row * HID_;
  for (int c = l; c < HID_; c += 32) o[c] = e[c];
}

__global__ void softmax_rows(float* __restrict__ data, int nrows, int ncols) {
  int row = blockIdx.x;
  if (row >= nrows) return;
  float* d = data + (long)row * ncols;
  __shared__ float red[32];
  int tid = threadIdx.x, l = tid & 31, w = tid >> 5;
  int nw = blockDim.x >> 5;
  float mx = -3.0e38f;
  for (int c = tid; c < ncols; c += blockDim.x) mx = fmaxf(mx, d[c]);
#pragma unroll
  for (int off = 16; off; off >>= 1) mx = fmaxf(mx, __shfl_xor(mx, off, 32));
  if (l == 0) red[w] = mx;
  __syncthreads();
  if (w == 0) {
    float v = (l < nw) ? red[l] : -3.0e38f;
#pragma unroll
    for (int off = 16; off; off >>= 1) v = fmaxf(v, __shfl_xor(v, off, 32));
    red[0] = v;
  }
  __syncthreads();
  mx = red[0];
  __syncthreads();
  float sum = 0.f;
  for (int c = tid; c < ncols; c += blockDim.x) {
    float e = __expf(d[c] - mx); d[c] = e; sum += e;
  }
#pragma unroll
  for (int off = 16; off; off >>= 1) sum += __shfl_xor(sum, off, 32);
  if (l == 0) red[w] = sum;
  __syncthreads();
  if (w == 0) {
    float v = (l < nw) ? red[l] : 0.f;
#pragma unroll
    for (int off = 16; off; off >>= 1) v += __shfl_xor(v, off, 32);
    red[0] = v;
  }
  __syncthreads();
  float inv = 1.f / red[0];
  for (int c = tid; c < ncols; c += blockDim.x) d[c] *= inv;
}

__global__ void logsoftmax_rows(float* __restrict__ data, int nrows, int ncols) {
  int row = blockIdx.x * 8 + (threadIdx.x >> 5);
  int l = threadIdx.x & 31;
  if (row >= nrows) return;
  float* d = data + (long)row * ncols;
  float mx = -3.0e38f;
  for (int c = l; c < ncols; c += 32) mx = fmaxf(mx, d[c]);
#pragma unroll
  for (int off = 16; off; off >>= 1) mx = fmaxf(mx, __shfl_xor(mx, off, 32));
  float s = 0.f;
  for (int c = l; c < ncols; c += 32) s += __expf(d[c] - mx);
#pragma unroll
  for (int off = 16; off; off >>= 1) s += __shfl_xor(s, off, 32);
  float ls = mx + __logf(s);
  for (int c = l; c < ncols; c += 32) d[c] -= ls;
}

// ---------------- host orchestration ----------------

static inline dim3 gemm_grid(int M, int N, int batch) {
  return dim3((N + 63) / 64, (M + 127) / 128, batch);
}

extern "C" void kernel_launch(void* const* d_in, const int* in_sizes, int n_in,
                              void* d_out, int out_size, void* d_ws, size_t ws_size,
                              hipStream_t stream) {
  (void)in_sizes; (void)n_in; (void)out_size; (void)ws_size;
  const float* inputs = (const float*)d_in[0];
  const float* lin1_W = (const float*)d_in[1];
  const float* lin1_b = (const float*)d_in[2];
  const float* Wih_f  = (const float*)d_in[3];
  const float* Whh_f  = (const float*)d_in[4];
  const float* b_f    = (const float*)d_in[5];
  const float* Wih_b  = (const float*)d_in[6];
  const float* Whh_b  = (const float*)d_in[7];
  const float* b_b    = (const float*)d_in[8];
  const float* lin2_W = (const float*)d_in[9];
  const float* lin2_b = (const float*)d_in[10];
  const float* vq_emb = (const float*)d_in[11];
  const float* Wq = (const float*)d_in[12];
  const float* bq = (const float*)d_in[13];
  const float* Wk = (const float*)d_in[14];
  const float* bk = (const float*)d_in[15];
  const float* Wv = (const float*)d_in[16];
  const float* bv = (const float*)d_in[17];
  const float* Wcls = (const float*)d_in[18];
  const float* bcls = (const float*)d_in[19];
  // d_in[20] input_lens, d_in[21] in_mask: all-valid in this configuration.

  float* out = (float*)d_out;
  const long ROWS = (long)B_ * L_;                 // 32000
  float* out_dec  = out;                           // [32000, 72]
  float* out_attn = out + ROWS * NCLS_;            // [32, 1000, 1000]
  float* out_ze   = out_attn + (long)B_ * L_ * L_; // [32000, 256]
  float* out_zq   = out_ze + ROWS * HID_;          // [32000, 256]

  // ---- workspace bump allocator ----
  char* ws = (char*)d_ws;
  size_t off = 0;
  auto alloc = [&](size_t bytes) -> void* {
    void* p = ws + off;
    off = (off + bytes + 255) & ~(size_t)255;
    return p;
  };
  u16* inputs_bf = (u16*)alloc(ROWS * INP_ * 2);
  u16* lin1W_bf  = (u16*)alloc((size_t)HID_ * INP_ * 2);
  u16* x_bf      = (u16*)alloc(ROWS * HID_ * 2);
  u16* WihF_bf   = (u16*)alloc((size_t)GATE_ * HID_ * 2);
  u16* WhhF_bf   = (u16*)alloc((size_t)GATE_ * HID_ * 2);
  u16* WihB_bf   = (u16*)alloc((size_t)GATE_ * HID_ * 2);
  u16* WhhB_bf   = (u16*)alloc((size_t)GATE_ * HID_ * 2);
  u16* hcat_bf   = (u16*)alloc(ROWS * 2 * HID_ * 2);
  u16* lin2W_bf  = (u16*)alloc((size_t)HID_ * 2 * HID_ * 2);
  u16* ze_bf     = (u16*)alloc(ROWS * HID_ * 2);
  u16* vqemb_bf  = (u16*)alloc((size_t)KVQ_ * HID_ * 2);
  float* vqnorm  = (float*)alloc(KVQ_ * 4);
  float* vqscore = (float*)alloc(ROWS * KVQ_ * 4);
  u16* zq_bf     = (u16*)alloc(ROWS * HID_ * 2);
  u16* Wq_bf     = (u16*)alloc((size_t)HID_ * HID_ * 2);
  u16* Wk_bf     = (u16*)alloc((size_t)HID_ * HID_ * 2);
  u16* Wv_bf     = (u16*)alloc((size_t)HID_ * HID_ * 2);
  u16* q_bf      = (u16*)alloc(ROWS * HID_ * 2);
  u16* k_bf      = (u16*)alloc(ROWS * HID_ * 2);
  u16* v_bf      = (u16*)alloc(ROWS * HID_ * 2);
  u16* attn_bf   = (u16*)alloc((size_t)B_ * L_ * 1024 * 2);
  u16* ctx_bf    = (u16*)alloc(ROWS * HID_ * 2);
  u16* Wcls_bf   = (u16*)alloc((size_t)NCLS_ * HID_ * 2);

  auto cvt = [&](const float* s, u16* d, int rows, int cols, int colsp) {
    long tot = (long)rows * colsp;
    cvt_pad<<<dim3((unsigned)((tot + 255) / 256)), 256, 0, stream>>>(s, d, rows, cols, colsp);
  };

  // ---- weight & input conversions (f32 -> bf16, K-padded) ----
  cvt(inputs, inputs_bf, (int)ROWS, IN_, INP_);
  cvt(lin1_W, lin1W_bf, HID_, IN_, INP_);
  cvt(Wih_f, WihF_bf, GATE_, HID_, HID_);
  cvt(Whh_f, WhhF_bf, GATE_, HID_, HID_);
  cvt(Wih_b, WihB_bf, GATE_, HID_, HID_);
  cvt(Whh_b, WhhB_bf, GATE_, HID_, HID_);
  cvt(lin2_W, lin2W_bf, HID_, 2 * HID_, 2 * HID_);
  cvt(vq_emb, vqemb_bf, KVQ_, HID_, HID_);
  cvt(Wq, Wq_bf, HID_, HID_, HID_);
  cvt(Wk, Wk_bf, HID_, HID_, HID_);
  cvt(Wv, Wv_bf, HID_, HID_, HID_);
  cvt(Wcls, Wcls_bf, NCLS_, HID_, HID_);
  vq_norm_k<<<2, 256, 0, stream>>>(vq_emb, vqnorm);

  // ---- 1) x = relu(inputs @ lin1_W^T + b) -> bf16 ----
  gemm_nt<<<gemm_grid((int)ROWS, HID_, 1), 256, 0, stream>>>(
      inputs_bf, lin1W_bf, nullptr, x_bf, lin1_b,
      (int)ROWS, HID_, INP_, INP_, INP_, HID_, 0, 0, 0, 1.f, 1);

  // ---- 2) bidirectional LSTM scan (2 WGs, gates via WMMA, c in VGPRs) ----
  lstm_scan<<<2, 1024, 0, stream>>>(x_bf, WihF_bf, WhhF_bf, b_f,
                                    WihB_bf, WhhB_bf, b_b, hcat_bf);

  // ---- 3) ze = hcat @ lin2_W^T + b -> d_out (f32) + bf16 ----
  gemm_nt<<<gemm_grid((int)ROWS, HID_, 1), 256, 0, stream>>>(
      hcat_bf, lin2W_bf, out_ze, ze_bf, lin2_b,
      (int)ROWS, HID_, 2 * HID_, 2 * HID_, 2 * HID_, HID_, 0, 0, 0, 1.f, 0);

  // ---- 4) VQ: scores = ze @ emb^T ; argmin(||e||^2 - 2s) ; zq = emb[idx] ----
  gemm_nt<<<gemm_grid((int)ROWS, KVQ_, 1), 256, 0, stream>>>(
      ze_bf, vqemb_bf, vqscore, nullptr, nullptr,
      (int)ROWS, KVQ_, HID_, HID_, HID_, KVQ_, 0, 0, 0, 1.f, 0);
  vq_select<<<(unsigned)((ROWS + 7) / 8), 256, 0, stream>>>(vqscore, vqnorm, vq_emb, out_zq);
  cvt(out_zq, zq_bf, (int)ROWS, HID_, HID_);      // dec_in == zq (forward)

  // ---- 5) q/k/v projections -> bf16 ----
  gemm_nt<<<gemm_grid((int)ROWS, HID_, 1), 256, 0, stream>>>(
      zq_bf, Wq_bf, nullptr, q_bf, bq, (int)ROWS, HID_, HID_, HID_, HID_, HID_, 0, 0, 0, 1.f, 0);
  gemm_nt<<<gemm_grid((int)ROWS, HID_, 1), 256, 0, stream>>>(
      zq_bf, Wk_bf, nullptr, k_bf, bk, (int)ROWS, HID_, HID_, HID_, HID_, HID_, 0, 0, 0, 1.f, 0);
  gemm_nt<<<gemm_grid((int)ROWS, HID_, 1), 256, 0, stream>>>(
      zq_bf, Wv_bf, nullptr, v_bf, bv, (int)ROWS, HID_, HID_, HID_, HID_, HID_, 0, 0, 0, 1.f, 0);

  // ---- 6) scores = q @ k^T / 16 (batched over B) -> softmax -> attn_w ----
  gemm_nt<<<gemm_grid(L_, L_, B_), 256, 0, stream>>>(
      q_bf, k_bf, out_attn, nullptr, nullptr,
      L_, L_, HID_, HID_, HID_, L_,
      (long)L_ * HID_, (long)L_ * HID_, (long)L_ * L_, 1.f / 16.f, 0);
  softmax_rows<<<(unsigned)ROWS, 256, 0, stream>>>(out_attn, (int)ROWS, L_);
  cvt(out_attn, attn_bf, (int)ROWS, L_, 1024);    // K padded 1000 -> 1024, zero fill

  // ---- 7) ctx = attn @ v (batched, NN form) -> bf16 ----
  gemm_nn<<<gemm_grid(L_, HID_, B_), 256, 0, stream>>>(
      attn_bf, v_bf, nullptr, ctx_bf, nullptr,
      L_, HID_, 1024, L_, 1024, HID_, HID_,
      (long)L_ * 1024, (long)L_ * HID_, (long)L_ * HID_, 1.f, 0);

  // ---- 8) logits = ctx @ Wcls^T + b ; log_softmax in place -> dec_out ----
  gemm_nt<<<gemm_grid((int)ROWS, NCLS_, 1), 256, 0, stream>>>(
      ctx_bf, Wcls_bf, out_dec, nullptr, bcls,
      (int)ROWS, NCLS_, HID_, HID_, HID_, NCLS_, 0, 0, 0, 1.f, 0);
  logsoftmax_rows<<<(unsigned)((ROWS + 7) / 8), 256, 0, stream>>>(out_dec, (int)ROWS, NCLS_);
}
